// EmbeddingPiGAN128_6296422056413
// MI455X (gfx1250) — matrix operand order repaired and checked by reading, not compiled, not run on hardware
//
#include <hip/hip_runtime.h>
#include <hip/hip_bf16.h>

// ---------------- constants from the reference ----------------
#define HID     128
#define ZDIM    100
#define GC      32
#define GD      96
#define BATCH   8
#define NPTS    65536
#define NLAYERS 8
#define MAPH    256
#define FDIM    ((NLAYERS + 1) * HID)   // 1152

typedef __attribute__((ext_vector_type(16))) _Float16     v16h;
typedef __attribute__((ext_vector_type(8)))  float        v8f;
typedef __attribute__((ext_vector_type(4)))  unsigned int u32x4;
typedef __attribute__((ext_vector_type(8)))  int          i32x8;
typedef __attribute__((ext_vector_type(4)))  int          i32x4;
typedef __attribute__((ext_vector_type(4)))  float        f32x4;

union F16Frag { u32x4 u[2]; v16h h; };

// Packed-weight sizes (halves)
#define MAIN_HALVES (8 * 8 * 4 * 512)     // 131072
#define COLOR_HALVES (8 * 5 * 512)        // 20480
#define PACK_HALVES (MAIN_HALVES + COLOR_HALVES)

// dynamic-LDS layout (bytes)
#define FILM_BYTES  (2 * FDIM * 4)                 // 9216
#define WBUF_OFF    FILM_BYTES                     // 9216
#define WBUF_BYTES  (COLOR_HALVES * 2)             // 40960
#define XBUF_OFF    (WBUF_OFF + WBUF_BYTES)        // 50176
#define XBUF_BYTES  (8 * 16 * 160 * 2)             // 40960
#define SMEM_BYTES  (XBUF_OFF + XBUF_BYTES)        // 91136

// ---------------- kernel 1: mapping network (tiny) ----------------
__global__ __launch_bounds__(MAPH) void map_kernel(
    const float* __restrict__ z,
    const float* __restrict__ w0, const float* __restrict__ b0,
    const float* __restrict__ w1, const float* __restrict__ b1,
    const float* __restrict__ w2, const float* __restrict__ b2,
    const float* __restrict__ w3, const float* __restrict__ b3,
    float* __restrict__ freqs, float* __restrict__ phases)
{
    __shared__ float zb[128];
    __shared__ float hA[MAPH];
    __shared__ float hB[MAPH];
    const int b = blockIdx.x;
    const int t = threadIdx.x;

    if (t < ZDIM) zb[t] = z[b * ZDIM + t];
    __syncthreads();

    float acc = b0[t];
    for (int k = 0; k < ZDIM; ++k) acc += zb[k] * w0[t * ZDIM + k];
    hA[t] = acc > 0.f ? acc : 0.2f * acc;
    __syncthreads();

    acc = b1[t];
    for (int k = 0; k < MAPH; ++k) acc += hA[k] * w1[t * MAPH + k];
    hB[t] = acc > 0.f ? acc : 0.2f * acc;
    __syncthreads();

    acc = b2[t];
    for (int k = 0; k < MAPH; ++k) acc += hB[k] * w2[t * MAPH + k];
    hA[t] = acc > 0.f ? acc : 0.2f * acc;
    __syncthreads();

    for (int o = t; o < 2 * FDIM; o += MAPH) {
        float a = b3[o];
        for (int k = 0; k < MAPH; ++k) a += hA[k] * w3[o * MAPH + k];
        if (o < FDIM) freqs[b * FDIM + o] = a * 15.f + 30.f;
        else          phases[b * FDIM + (o - FDIM)] = a;
    }
}

// ---------------- kernel 2: pack weights into WMMA B-fragment layout ----------
__global__ __launch_bounds__(256) void pack_kernel(
    const float* __restrict__ net_w0,   // (128, 35)
    const float* __restrict__ net_ws,   // (7, 128, 128)
    const float* __restrict__ color_w,  // (128, 131)
    _Float16* __restrict__ packed)
{
    int idx = blockIdx.x * 256 + threadIdx.x;
    if (idx >= PACK_HALVES) return;

    if (idx < MAIN_HALVES) {
        int L  = idx / 16384;  int r  = idx % 16384;
        int n  = r   / 2048;   int r2 = r   % 2048;
        int kb = r2  / 512;    int r3 = r2  % 512;
        int lane = r3 / 16;    int h  = r3  % 16;
        int row = n * 16 + (lane & 15);
        int K   = kb * 32 + (lane >> 4) * 16 + h;
        float v;
        if (L == 0) v = (K < GC + 3) ? net_w0[row * (GC + 3) + K] : 0.f;
        else        v = net_ws[(L - 1) * HID * HID + row * HID + K];
        packed[idx] = (_Float16)v;
    } else {
        int r  = idx - MAIN_HALVES;
        int n  = r  / (5 * 512); int r2 = r  % (5 * 512);
        int kb = r2 / 512;       int r3 = r2 % 512;
        int lane = r3 / 16;      int h  = r3 % 16;
        int row = n * 16 + (lane & 15);
        int K   = kb * 32 + (lane >> 4) * 16 + h;
        float v = 0.f;
        if (K < 128)      v = color_w[row * 131 + 3 + K];
        else if (K < 131) v = color_w[row * 131 + (K - 128)];
        packed[idx] = (_Float16)v;
    }
}

// ---------------- TDM staging: contiguous block global -> LDS -----------------
// D# per ISA §8.3/8.4: 2D tile, data_size=2B, tile_dim0 = nHalves (<=65535),
// tile_dim1 = 1, tensor_dim0 = stride = nHalves. Issued by one wave, tracked
// with TENSORcnt.
__device__ __attribute__((always_inline)) inline void tdm_load_lds(
    const _Float16* gsrc, unsigned ldsByteOff, unsigned nHalves)
{
    const unsigned long long ga = (unsigned long long)(uintptr_t)gsrc;
    const unsigned lds = __builtin_amdgcn_groupstaticsize() + ldsByteOff;
    u32x4 g0 = { 1u,                                        // count=1, user D#
                 lds,                                       // lds_addr
                 (unsigned)(ga & 0xFFFFFFFFu),              // global_addr lo
                 (unsigned)((ga >> 32) & 0x1FFFFFFu) | (2u << 30) }; // hi | type=2
    i32x8 g1 = { (int)(1u << 16),                // data_size=1 (2 bytes), mask=0
                 (int)((nHalves & 0xFFFFu) << 16),  // tensor_dim0[15:0] @63:48
                 (int)(1u << 16),                // tensor_dim1=1 @95:80
                 (int)((nHalves & 0xFFFFu) << 16),  // tile_dim0 @127:112
                 1,                              // tile_dim1=1, tile_dim2=0
                 (int)nHalves,                   // tensor_dim0_stride[31:0]
                 0, 0 };
    i32x4 gz = {0, 0, 0, 0};
#if __clang_major__ >= 23
    i32x8 gz8 = {0, 0, 0, 0, 0, 0, 0, 0};
    __builtin_amdgcn_tensor_load_to_lds(g0, g1, gz, gz, gz8, 0);
#else
    __builtin_amdgcn_tensor_load_to_lds(g0, g1, gz, gz, 0);
#endif
}

// ---------------- FiLM layer: WMMA matmul + sin activation ----------------
template<int KB>
__device__ __attribute__((always_inline)) inline void layer_op(
    const _Float16* __restrict__ wbuf, _Float16* xbuf,
    int wv, int lane,
    const float* __restrict__ fF, const float* __restrict__ fP,
    const float* __restrict__ bias)
{
    const int row = lane & 15;
    const int g   = lane >> 4;
    const _Float16* xr = xbuf + (wv * 16 + row) * 160;

    F16Frag a[KB];
#pragma unroll
    for (int kb = 0; kb < KB; ++kb) {
        a[kb].u[0] = *(const u32x4*)(xr + kb * 32 + g * 16);
        a[kb].u[1] = *(const u32x4*)(xr + kb * 32 + g * 16 + 8);
    }

#pragma unroll
    for (int n = 0; n < 8; ++n) {
        v8f acc = {0.f, 0.f, 0.f, 0.f, 0.f, 0.f, 0.f, 0.f};
#pragma unroll
        for (int kb = 0; kb < KB; ++kb) {
            F16Frag bf;
            const _Float16* wp = wbuf + ((n * KB + kb) * 32 + lane) * 16;
            bf.u[0] = *(const u32x4*)wp;
            bf.u[1] = *(const u32x4*)(wp + 8);
            acc = __builtin_amdgcn_wmma_f32_16x16x32_f16(
                false, a[kb].h, false, bf.h, (short)0, acc, false, false);
        }
        const int feat = n * 16 + row;
        const float bs = bias[feat];
        const float fr = fF[feat];
        const float ph = fP[feat];
        _Float16* xw = xbuf + (wv * 16 + 8 * g) * 160 + feat;
#pragma unroll
        for (int i = 0; i < 8; ++i) {
            float v = __sinf(fr * (acc[i] + bs) + ph);
            xw[i * 160] = (_Float16)v;
        }
    }
}

// ---------------- kernel 3: main per-point pipeline ----------------
__global__ __launch_bounds__(256) void pigan_main(
    const float* __restrict__ input,      // (B,N,3)
    const float* __restrict__ rays,       // (B,N,3)
    const float* __restrict__ grid,       // (32,96,96,96)
    const float* __restrict__ net_b0,     // (128)
    const float* __restrict__ net_bs,     // (7,128)
    const float* __restrict__ final_w,    // (1,128)
    const float* __restrict__ final_b,    // (1)
    const float* __restrict__ color_b,    // (128)
    const float* __restrict__ clin_w,     // (3,128)
    const float* __restrict__ clin_b,     // (3)
    const float* __restrict__ freqs,      // (B,1152) ws
    const float* __restrict__ phases,     // (B,1152) ws
    const _Float16* __restrict__ packedW, // ws
    float* __restrict__ out)              // (B,N,4)
{
    extern __shared__ char smem[];
    float*    filmF = (float*)smem;                      // 1152 f32
    float*    filmP = filmF + FDIM;                      // 1152 f32
    _Float16* wbuf  = (_Float16*)(smem + WBUF_OFF);      // 20480 halves
    _Float16* xbuf  = (_Float16*)(smem + XBUF_OFF);      // 8*16*160 halves

    const int b    = blockIdx.y;
    const int tid  = threadIdx.x;
    const int lane = tid & 31;
    const int wv   = tid >> 5;
    const int n0   = blockIdx.x * 128 + wv * 16;

    for (int i = tid; i < FDIM; i += 256) {
        filmF[i] = freqs[b * FDIM + i];
        filmP[i] = phases[b * FDIM + i];
    }

    // ---- grid sample (lane = channel), build layer-0 input in xbuf ----
    for (int p = 0; p < 16; ++p) {
        const size_t nIdx = (size_t)b * NPTS + (n0 + p);
        const float cx = input[nIdx * 3 + 0] * (2.f / 0.24f);
        const float cy = input[nIdx * 3 + 1] * (2.f / 0.24f);
        const float cz = input[nIdx * 3 + 2] * (2.f / 0.24f);
        const float fx = (cx + 1.f) * 0.5f * (GD - 1);
        const float fy = (cy + 1.f) * 0.5f * (GD - 1);
        const float fz = (cz + 1.f) * 0.5f * (GD - 1);
        const float x0f = floorf(fx), y0f = floorf(fy), z0f = floorf(fz);
        const int x0i = (int)x0f, y0i = (int)y0f, z0i = (int)z0f;

        float acc = 0.f;
#pragma unroll
        for (int cor = 0; cor < 8; ++cor) {
            const int dx = cor & 1, dy = (cor >> 1) & 1, dz = cor >> 2;
            const int xi = x0i + dx, yi = y0i + dy, zi = z0i + dz;
            const float wx = dx ? (fx - x0f) : (x0f + 1.f - fx);
            const float wy = dy ? (fy - y0f) : (y0f + 1.f - fy);
            const float wz = dz ? (fz - z0f) : (z0f + 1.f - fz);
            const bool valid = (xi >= 0) & (xi < GD) & (yi >= 0) & (yi < GD) &
                               (zi >= 0) & (zi < GD);
            const int xc = min(max(xi, 0), GD - 1);
            const int yc = min(max(yi, 0), GD - 1);
            const int zc = min(max(zi, 0), GD - 1);
            const float gv = grid[((lane * GD + zc) * GD + yc) * GD + xc];
            acc += valid ? gv * wx * wy * wz : 0.f;
        }
        _Float16* xr = xbuf + (wv * 16 + p) * 160;
        xr[lane] = (_Float16)acc;                               // feat 0..31
        if (lane < 3) {
            const float cv = (lane == 0) ? cx : (lane == 1) ? cy : cz;
            xr[32 + lane]  = (_Float16)cv;                      // feat 32..34
            xr[128 + lane] = (_Float16)rays[nIdx * 3 + lane];   // color K 128..130
        }
        for (int c = 35 + lane; c < 128; c += 32) xr[c] = (_Float16)0.f;
        for (int c = 131 + lane; c < 160; c += 32) xr[c] = (_Float16)0.f;
    }

    float sigmaV = 0.f;

    // ---- 8 main FiLM layers (K=128 padded), weights staged by TDM ----
    for (int L = 0; L < 8; ++L) {
        __syncthreads();                       // WAR: previous wbuf readers done
        if (wv == 0) {
            tdm_load_lds(packedW + L * 16384, WBUF_OFF, 16384);
            __builtin_amdgcn_s_wait_tensorcnt(0);
        }
        __syncthreads();                       // RAW: wbuf visible to all waves
        layer_op<4>(wbuf, xbuf, wv, lane,
                    filmF + L * HID, filmP + L * HID,
                    (L == 0) ? net_b0 : (net_bs + (L - 1) * HID));
        if (L == 7) {
            if (lane < 16) {                   // sigma head before color overwrite
                float s = final_b[0];
                const _Float16* xr = xbuf + (wv * 16 + lane) * 160;
                for (int k = 0; k < HID; ++k) s += (float)xr[k] * final_w[k];
                sigmaV = s;
            }
        }
    }

    // ---- color FiLM layer (K=160 padded: x | rd | 0) ----
    __syncthreads();
    if (wv == 0) {
        tdm_load_lds(packedW + MAIN_HALVES, WBUF_OFF, COLOR_HALVES);
        __builtin_amdgcn_s_wait_tensorcnt(0);
    }
    __syncthreads();
    layer_op<5>(wbuf, xbuf, wv, lane, filmF + 8 * HID, filmP + 8 * HID, color_b);

    // ---- clin head + sigmoid + store ----
    if (lane < 16) {
        const _Float16* xr = xbuf + (wv * 16 + lane) * 160;
        float r = clin_b[0], g = clin_b[1], bl = clin_b[2];
        for (int k = 0; k < HID; ++k) {
            const float v = (float)xr[k];
            r  += v * clin_w[k];
            g  += v * clin_w[HID + k];
            bl += v * clin_w[2 * HID + k];
        }
        const float sr = 1.f / (1.f + __expf(-r));
        const float sg = 1.f / (1.f + __expf(-g));
        const float sb = 1.f / (1.f + __expf(-bl));
        f32x4 res = {sr, sg, sb, sigmaV};
        *(f32x4*)(out + ((size_t)b * NPTS + n0 + lane) * 4) = res;
    }
}

// ---------------- launcher ----------------
extern "C" void kernel_launch(void* const* d_in, const int* in_sizes, int n_in,
                              void* d_out, int out_size, void* d_ws, size_t ws_size,
                              hipStream_t stream) {
    const float* input  = (const float*)d_in[0];
    const float* z      = (const float*)d_in[1];
    const float* rays   = (const float*)d_in[2];
    const float* grid   = (const float*)d_in[3];
    const float* map_w0 = (const float*)d_in[4];
    const float* map_b0 = (const float*)d_in[5];
    const float* map_w1 = (const float*)d_in[6];
    const float* map_b1 = (const float*)d_in[7];
    const float* map_w2 = (const float*)d_in[8];
    const float* map_b2 = (const float*)d_in[9];
    const float* map_w3 = (const float*)d_in[10];
    const float* map_b3 = (const float*)d_in[11];
    const float* net_w0 = (const float*)d_in[12];
    const float* net_b0 = (const float*)d_in[13];
    const float* net_ws = (const float*)d_in[14];
    const float* net_bs = (const float*)d_in[15];
    const float* final_w= (const float*)d_in[16];
    const float* final_b= (const float*)d_in[17];
    const float* color_w= (const float*)d_in[18];
    const float* color_b= (const float*)d_in[19];
    const float* clin_w = (const float*)d_in[20];
    const float* clin_b = (const float*)d_in[21];

    float*    ws_freqs  = (float*)d_ws;                          // 8*1152
    float*    ws_phases = ws_freqs + BATCH * FDIM;               // 8*1152
    _Float16* ws_packed = (_Float16*)(ws_phases + BATCH * FDIM); // PACK_HALVES

    map_kernel<<<BATCH, MAPH, 0, stream>>>(
        z, map_w0, map_b0, map_w1, map_b1, map_w2, map_b2, map_w3, map_b3,
        ws_freqs, ws_phases);

    pack_kernel<<<(PACK_HALVES + 255) / 256, 256, 0, stream>>>(
        net_w0, net_ws, color_w, ws_packed);

    dim3 gdim(NPTS / 128, BATCH);
    pigan_main<<<gdim, 256, SMEM_BYTES, stream>>>(
        input, rays, grid, net_b0, net_bs, final_w, final_b,
        color_b, clin_w, clin_b, ws_freqs, ws_phases, ws_packed,
        (float*)d_out);
}
